// TransformerModel_48421461295484
// MI455X (gfx1250) — matrix operand order repaired and checked by reading, not compiled, not run on hardware
//
#include <hip/hip_runtime.h>

// ---------------------------------------------------------------------------
// MI455X (gfx1250) transformer forward pass.
// All heavy math goes through V_WMMA_F32_16X16X32_BF16 (wave32, f32 accum).
// fp32 operands are converted to bf16 (RNE) while being staged into LDS;
// each weight matrix is read exactly once per forward pass so on-the-fly
// conversion is bandwidth-optimal (HBM-bound workload, ~0.95 TFLOP vs
// ~1.5 GB of unique traffic at 23.3 TB/s).
//
// LDS layouts are chosen so every WMMA fragment is loaded with exactly two
// DS_LOAD_B128s (ISA 7.12.2: 16-bit A fragment = two contiguous 8-elem runs;
// B fragment in n-major LDS = likewise two 8-elem runs). Leading dims are
// 72 elements (144 B) so row starts hit distinct 4-bank-aligned offsets
// (36*i mod 64) -> conflict-free b128 access.
// ---------------------------------------------------------------------------

typedef __bf16 bf16_t;
typedef bf16_t v16bf __attribute__((ext_vector_type(16)));
typedef float  v8f   __attribute__((ext_vector_type(8)));
typedef unsigned int v4u __attribute__((ext_vector_type(4)));

union FragBF { v16bf v; v4u q[2]; };

__device__ __forceinline__ unsigned short f2bf(float f) {
    union { float f; unsigned int u; } c; c.f = f;
    unsigned int u = c.u;
    unsigned int r = (u + 0x7FFFu + ((u >> 16) & 1u)) >> 16;   // round-nearest-even
    return (unsigned short)r;
}
__device__ __forceinline__ unsigned int pack_bf2(float lo, float hi) {
    return ((unsigned int)f2bf(hi) << 16) | (unsigned int)f2bf(lo);
}

#define TILE_M 128
#define TILE_N 64
#define KCHUNK 64          // two 32-deep WMMA sub-chunks per barrier interval
#define A_LD   72          // bf16 elements per LDS row (144 B, 16B-aligned, conflict-free)
#define B_LD   72

// ---------------------------------------------------------------------------
// Generic batched GEMM: C = act(alpha * A @ B(^T) + bias) + residual
//   A: [M,K] row-major (lda); B: transB==0 -> [K,N] (ldb), transB==1 -> [N,K]
//   Batch offset: z=(zo*batchDiv+zi) -> off = zo*s?o + zi*s?i
//   Requires M%128==0, N%64==0, K%64==0 (true for every GEMM in this model).
// 256 threads = 8 waves (4x2); wave tile 32x32 = 2x2 WMMA 16x16x32 fragments.
// ---------------------------------------------------------------------------
__global__ void __launch_bounds__(256)
gemm_wmma_kernel(const float* __restrict__ A, long long lda,
                 const float* __restrict__ B, long long ldb,
                 float* __restrict__ C, long long ldc,
                 int M, int N, int K,
                 const float* __restrict__ bias,
                 const float* __restrict__ residual, long long ldr,
                 float alpha, int relu, int transB,
                 int batchDiv,
                 long long sAo, long long sAi,
                 long long sBo, long long sBi,
                 long long sCo, long long sCi)
{
    __shared__ __align__(16) unsigned short As[TILE_M * A_LD];   // 18432 B
    __shared__ __align__(16) unsigned short Bs[TILE_N * B_LD];   //  9216 B (n-major)

    const int tid  = threadIdx.x;
    const int lane = tid & 31;
    const int wave = tid >> 5;
    const int wm   = wave & 3;     // wave row group (0..3)
    const int wn   = wave >> 2;    // wave col group (0..1)

    const int bz = blockIdx.z;
    const int zo = bz / batchDiv;
    const int zi = bz - zo * batchDiv;
    const long long offA = (long long)zo * sAo + (long long)zi * sAi;
    const long long offB = (long long)zo * sBo + (long long)zi * sBi;
    const long long offC = (long long)zo * sCo + (long long)zi * sCi;

    const int m0 = blockIdx.y * TILE_M;
    const int n0 = blockIdx.x * TILE_N;

    v8f acc[2][2] = {};

    for (int kk = 0; kk < K; kk += KCHUNK) {
        // ---- prefetch next K-chunk (global_prefetch_b8, speculative) ----
        if (kk + KCHUNK < K) {
            __builtin_prefetch(A + offA + (long long)(m0 + (tid >> 1)) * lda
                                 + kk + KCHUNK + ((tid & 1) << 5), 0, 1);
            if (!transB)
                __builtin_prefetch(B + offB + (long long)(kk + KCHUNK + (tid >> 3)) * ldb
                                     + n0 + ((tid & 7) << 3), 0, 1);
            else
                __builtin_prefetch(B + offB + (long long)(n0 + (tid >> 2)) * ldb
                                     + kk + KCHUNK + ((tid & 3) << 4), 0, 1);
        }

        // ---- stage A tile (128x64 fp32 -> bf16 pairs), float2 loads ----
        #pragma unroll
        for (int idx = tid; idx < TILE_M * (KCHUNK / 2); idx += 256) {   // 16 iters
            int r = idx >> 5, c2 = idx & 31;
            float2 f = *(const float2*)(A + offA + (long long)(m0 + r) * lda
                                          + kk + 2 * c2);
            *(unsigned int*)&As[r * A_LD + 2 * c2] = pack_bf2(f.x, f.y);
        }
        // ---- stage B tile into n-major LDS (64 n x 64 k) ----
        if (!transB) {
            #pragma unroll
            for (int idx = tid; idx < (KCHUNK / 2) * TILE_N; idx += 256) { // 8 iters
                int r2 = idx >> 6, c = idx & 63;
                const float* bp = B + offB + (long long)(kk + 2 * r2) * ldb + (n0 + c);
                *(unsigned int*)&Bs[c * B_LD + 2 * r2] = pack_bf2(bp[0], bp[ldb]);
            }
        } else {
            #pragma unroll
            for (int idx = tid; idx < TILE_N * (KCHUNK / 2); idx += 256) { // 8 iters
                int c = idx >> 5, k2 = idx & 31;
                float2 f = *(const float2*)(B + offB + (long long)(n0 + c) * ldb
                                              + kk + 2 * k2);
                *(unsigned int*)&Bs[c * B_LD + 2 * k2] = pack_bf2(f.x, f.y);
            }
        }
        __syncthreads();

        // ---- 2 sub-chunks x 4 fragments -> 8 WMMA per barrier interval ----
        #pragma unroll
        for (int sub = 0; sub < 2; ++sub) {
            const int ksub = sub * 32;
            FragBF fa[2], fb[2];
            // A fragment: elements e<8 -> k=khalf+e ; e>=8 -> k=16+khalf+(e-8)
            const int khalf = ((lane >> 4) << 3) + ksub;
            const int mbase = wm * 32 + (lane & 15);
            #pragma unroll
            for (int tm = 0; tm < 2; ++tm) {
                int mr = mbase + tm * 16;
                fa[tm].q[0] = *(const v4u*)&As[mr * A_LD + khalf];
                fa[tm].q[1] = *(const v4u*)&As[mr * A_LD + khalf + 16];
            }
            // B fragment: elements e<8 -> k=kbase+e ; e>=8 -> k=16+kbase+(e-8)
            const int kbase = ((lane >> 4) << 3) + ksub;
            const int nbase = wn * 32 + (lane & 15);
            #pragma unroll
            for (int tn = 0; tn < 2; ++tn) {
                int nc = nbase + tn * 16;
                fb[tn].q[0] = *(const v4u*)&Bs[nc * B_LD + kbase];
                fb[tn].q[1] = *(const v4u*)&Bs[nc * B_LD + kbase + 16];
            }
            #pragma unroll
            for (int tm = 0; tm < 2; ++tm)
                #pragma unroll
                for (int tn = 0; tn < 2; ++tn)
                    acc[tm][tn] = __builtin_amdgcn_wmma_f32_16x16x32_bf16(
                        false, fa[tm].v, false, fb[tn].v,
                        (short)0, acc[tm][tn], false, false);
        }
        __syncthreads();
    }

    // ---- epilogue: alpha, bias, relu, residual ----
    #pragma unroll
    for (int tm = 0; tm < 2; ++tm)
        #pragma unroll
        for (int tn = 0; tn < 2; ++tn)
            #pragma unroll
            for (int r = 0; r < 8; ++r) {
                int m = m0 + wm * 32 + tm * 16 + r + ((lane >> 4) << 3);
                int n = n0 + wn * 32 + tn * 16 + (lane & 15);
                float v = acc[tm][tn][r] * alpha;
                if (bias)     v += bias[n];
                if (relu)     v = fmaxf(v, 0.0f);
                if (residual) v += residual[(long long)m * ldr + n];
                C[offC + (long long)m * ldc + n] = v;
            }
}

// ---------------------------------------------------------------------------
// Embedding gather * sqrt(D) + sinusoidal positional encoding
// ---------------------------------------------------------------------------
__global__ void __launch_bounds__(256)
embed_kernel(const float* __restrict__ emb, const int* __restrict__ ids,
             float* __restrict__ out, int T, int D, float scale)
{
    int row = blockIdx.x;             // n*T + t
    int t   = row % T;
    int id  = ids[row];
    const float* e = emb + (long long)id * D;
    float* o = out + (long long)row * D;
    int half = D >> 1;
    float c = 9.2103403719761836f / (float)(half - 1);   // ln(10000)/(half-1)
    for (int d = threadIdx.x; d < D; d += 256) {
        int dd = (d < half) ? d : d - half;
        float inv = __expf(-(float)dd * c);
        float ang = (float)t * inv;
        float pe  = (d < half) ? __sinf(ang) : __cosf(ang);
        o[d] = e[d] * scale + pe;
    }
}

// ---------------------------------------------------------------------------
// LayerNorm over last dim (eps = 1e-3, keras default)
// ---------------------------------------------------------------------------
__global__ void __launch_bounds__(256)
layernorm_kernel(const float* __restrict__ x, const float* __restrict__ g,
                 const float* __restrict__ b, float* __restrict__ out, int D)
{
    __shared__ float s1[256], s2[256];
    long long row = blockIdx.x;
    const float* xr = x + row * D;
    float sum = 0.f, sq = 0.f;
    for (int d = threadIdx.x; d < D; d += 256) { float v = xr[d]; sum += v; sq += v * v; }
    s1[threadIdx.x] = sum; s2[threadIdx.x] = sq;
    __syncthreads();
    for (int s = 128; s > 0; s >>= 1) {
        if (threadIdx.x < s) { s1[threadIdx.x] += s1[threadIdx.x + s];
                               s2[threadIdx.x] += s2[threadIdx.x + s]; }
        __syncthreads();
    }
    float mean = s1[0] / (float)D;
    float var  = s2[0] / (float)D - mean * mean;
    float inv  = rsqrtf(var + 1e-3f);
    float* orow = out + row * D;
    for (int d = threadIdx.x; d < D; d += 256)
        orow[d] = (xr[d] - mean) * inv * g[d] + b[d];
}

// ---------------------------------------------------------------------------
// Masked softmax over last dim T (=512). mode 0: source-pad mask via src_ids;
// mode 1: causal. 128 threads, 4 elements each.
// ---------------------------------------------------------------------------
__global__ void __launch_bounds__(128)
softmax_mask_kernel(float* __restrict__ sc, const int* __restrict__ src_ids,
                    int mode, int T, int H)
{
    __shared__ float red[128];
    int row = blockIdx.x;              // n*H*T + h*T + q
    int q   = row % T;
    int n   = row / (T * H);
    float* p = sc + (long long)row * T;
    const int* src = (mode == 0) ? (src_ids + n * T) : nullptr;
    float vals[4];
    float mx = -3.4e38f;
    #pragma unroll
    for (int i = 0; i < 4; ++i) {
        int r = threadIdx.x + i * 128;
        float v = p[r];
        bool masked = (mode == 1) ? (r > q) : (src[r] == 0);
        if (masked) v += -1.0e9f;
        vals[i] = v;
        mx = fmaxf(mx, v);
    }
    red[threadIdx.x] = mx; __syncthreads();
    for (int s = 64; s > 0; s >>= 1) {
        if (threadIdx.x < s) red[threadIdx.x] = fmaxf(red[threadIdx.x], red[threadIdx.x + s]);
        __syncthreads();
    }
    mx = red[0]; __syncthreads();
    float sum = 0.f;
    #pragma unroll
    for (int i = 0; i < 4; ++i) { vals[i] = __expf(vals[i] - mx); sum += vals[i]; }
    red[threadIdx.x] = sum; __syncthreads();
    for (int s = 64; s > 0; s >>= 1) {
        if (threadIdx.x < s) red[threadIdx.x] += red[threadIdx.x + s];
        __syncthreads();
    }
    float inv = 1.0f / red[0];
    #pragma unroll
    for (int i = 0; i < 4; ++i) p[threadIdx.x + i * 128] = vals[i] * inv;
}

// ---------------------------------------------------------------------------
// Host orchestration
// ---------------------------------------------------------------------------
extern "C" void kernel_launch(void* const* d_in, const int* in_sizes, int n_in,
                              void* d_out, int out_size, void* d_ws, size_t ws_size,
                              hipStream_t stream) {
    (void)in_sizes; (void)n_in; (void)out_size; (void)ws_size;
    using ll = long long;

    constexpr int NB = 4, T = 512, V = 32000, D = 1024, H = 16, F = 4096, NL = 6, S = 64;
    constexpr int NT = NB * T;                       // 2048 tokens per stream
    constexpr ll  DD = (ll)D * D;                    // per-layer attn weight stride

    const float* emb   = (const float*)d_in[0];
    const float* e_wq  = (const float*)d_in[1];
    const float* e_wk  = (const float*)d_in[2];
    const float* e_wv  = (const float*)d_in[3];
    const float* e_wo  = (const float*)d_in[4];
    const float* e_l1g = (const float*)d_in[5];
    const float* e_l1b = (const float*)d_in[6];
    const float* e_w1  = (const float*)d_in[7];
    const float* e_b1  = (const float*)d_in[8];
    const float* e_w2  = (const float*)d_in[9];
    const float* e_b2  = (const float*)d_in[10];
    const float* e_l2g = (const float*)d_in[11];
    const float* e_l2b = (const float*)d_in[12];
    const float* e_lfg = (const float*)d_in[13];
    const float* e_lfb = (const float*)d_in[14];
    const float* d_wq1 = (const float*)d_in[15];
    const float* d_wk1 = (const float*)d_in[16];
    const float* d_wv1 = (const float*)d_in[17];
    const float* d_wo1 = (const float*)d_in[18];
    const float* d_l1g = (const float*)d_in[19];
    const float* d_l1b = (const float*)d_in[20];
    const float* d_wq2 = (const float*)d_in[21];
    const float* d_wk2 = (const float*)d_in[22];
    const float* d_wv2 = (const float*)d_in[23];
    const float* d_wo2 = (const float*)d_in[24];
    const float* d_l2g = (const float*)d_in[25];
    const float* d_l2b = (const float*)d_in[26];
    const float* d_w1  = (const float*)d_in[27];
    const float* d_b1  = (const float*)d_in[28];
    const float* d_w2  = (const float*)d_in[29];
    const float* d_b2  = (const float*)d_in[30];
    const float* d_l3g = (const float*)d_in[31];
    const float* d_l3b = (const float*)d_in[32];
    const float* d_lfg = (const float*)d_in[33];
    const float* d_lfb = (const float*)d_in[34];
    const int*   src_ids = (const int*)d_in[35];
    const int*   tgt_ids = (const int*)d_in[36];

    // workspace carve-up (floats): ~152 MB total
    float* ws   = (float*)d_ws;
    float* x    = ws;                                   // [NT,D] activations
    float* encO = x    + (size_t)NT * D;                // [NT,D] encoder output
    float* hbuf = encO + (size_t)NT * D;                // [NT,D] LN output
    float* qb   = hbuf + (size_t)NT * D;                // [N,T,H,S]
    float* kb   = qb   + (size_t)NT * D;
    float* vb   = kb   + (size_t)NT * D;
    float* ob   = vb   + (size_t)NT * D;                // attn out pre-merge
    float* sc   = ob   + (size_t)NT * D;                // [N,H,T,T] 64 MB
    float* f1b  = sc   + (size_t)NB * H * T * T;        // [NT,F]   32 MB

    // plain GEMM (batch=1). residual shares ldc with C in every use here.
    auto G = [&](const float* A, ll lda, const float* B, ll ldb,
                 float* C, ll ldc, int M, int Nn, int K,
                 const float* bias, const float* res,
                 float alpha, int relu, int tB) {
        dim3 grid(Nn / TILE_N, M / TILE_M, 1);
        gemm_wmma_kernel<<<grid, 256, 0, stream>>>(A, lda, B, ldb, C, ldc, M, Nn, K,
            bias, res, ldc, alpha, relu, tB, 1, 0, 0, 0, 0, 0, 0);
    };
    // batched GEMM over (n,h)
    auto GB = [&](const float* A, ll lda, ll sAo, ll sAi,
                  const float* B, ll ldb, ll sBo, ll sBi,
                  float* C, ll ldc, ll sCo, ll sCi,
                  int M, int Nn, int K, int tB) {
        dim3 grid(Nn / TILE_N, M / TILE_M, NB * H);
        gemm_wmma_kernel<<<grid, 256, 0, stream>>>(A, lda, B, ldb, C, ldc, M, Nn, K,
            nullptr, nullptr, 0, 1.0f, 0, tB, H, sAo, sAi, sBo, sBi, sCo, sCi);
    };
    auto LN = [&](const float* in, const float* g, const float* b, float* out) {
        layernorm_kernel<<<NT, 256, 0, stream>>>(in, g, b, out, D);
    };
    // full multi-head attention: xio += attn(q_in, kv_in)
    auto ATTN = [&](const float* q_in, const float* kv_in,
                    const float* wq, const float* wk, const float* wv, const float* wo,
                    int mode, const int* mask_ids, float* xio) {
        G(q_in,  D, wq, D, qb, D, NT, D, D, nullptr, nullptr, 0.125f /* S^-0.5 */, 0, 0);
        G(kv_in, D, wk, D, kb, D, NT, D, D, nullptr, nullptr, 1.0f, 0, 0);
        G(kv_in, D, wv, D, vb, D, NT, D, D, nullptr, nullptr, 1.0f, 0, 0);
        // sims[n,h,q,r] = Q[n,q,h,:] . K[n,r,h,:]   (A@B^T per (n,h))
        GB(qb, D, (ll)T * D, S,  kb, D, (ll)T * D, S,
           sc, T, (ll)H * T * T, (ll)T * T,  T, T, S, 1);
        softmax_mask_kernel<<<NB * H * T, 128, 0, stream>>>(sc, mask_ids, mode, T, H);
        // o[n,q,h,s] = sum_r P[n,h,q,r] * V[n,r,h,s]
        GB(sc, T, (ll)H * T * T, (ll)T * T,  vb, D, (ll)T * D, S,
           ob, D, (ll)T * D, S,  T, S, T, 0);
        // merge heads + residual: xio = xio + o @ wo
        G(ob, D, wo, D, xio, D, NT, D, D, nullptr, xio, 1.0f, 0, 0);
    };
    auto FFN = [&](const float* w1, const float* b1, const float* w2, const float* b2,
                   float* xio) {
        G(hbuf, D, w1, F, f1b, F, NT, F, D, b1, nullptr, 1.0f, /*relu*/1, 0);
        G(f1b,  F, w2, D, xio, D, NT, D, F, b2, xio,     1.0f, 0, 0);
    };

    const float scale = 32.0f;   // sqrt(D)

    // ------------------------- encoder -------------------------
    embed_kernel<<<NT, 256, 0, stream>>>(emb, src_ids, x, T, D, scale);
    for (int i = 0; i < NL; ++i) {
        LN(x, e_l1g + (ll)i * D, e_l1b + (ll)i * D, hbuf);
        ATTN(hbuf, hbuf, e_wq + i * DD, e_wk + i * DD, e_wv + i * DD, e_wo + i * DD,
             /*pad mask*/0, src_ids, x);
        LN(x, e_l2g + (ll)i * D, e_l2b + (ll)i * D, hbuf);
        FFN(e_w1 + (ll)i * D * F, e_b1 + (ll)i * F,
            e_w2 + (ll)i * F * D, e_b2 + (ll)i * D, x);
    }
    LN(x, e_lfg, e_lfb, encO);

    // ------------------------- decoder -------------------------
    embed_kernel<<<NT, 256, 0, stream>>>(emb, tgt_ids, x, T, D, scale);
    for (int i = 0; i < NL; ++i) {
        LN(x, d_l1g + (ll)i * D, d_l1b + (ll)i * D, hbuf);
        ATTN(hbuf, hbuf, d_wq1 + i * DD, d_wk1 + i * DD, d_wv1 + i * DD, d_wo1 + i * DD,
             /*causal*/1, nullptr, x);
        LN(x, d_l2g + (ll)i * D, d_l2b + (ll)i * D, hbuf);
        ATTN(hbuf, encO, d_wq2 + i * DD, d_wk2 + i * DD, d_wv2 + i * DD, d_wo2 + i * DD,
             /*pad mask*/0, src_ids, x);
        LN(x, d_l3g + (ll)i * D, d_l3b + (ll)i * D, hbuf);
        FFN(d_w1 + (ll)i * D * F, d_b1 + (ll)i * F,
            d_w2 + (ll)i * F * D, d_b2 + (ll)i * D, x);
    }
    LN(x, d_lfg, d_lfb, hbuf);

    // logits = dec_out @ emb^T  (B = emb [V,D] row-major, transB)
    G(hbuf, D, emb, D, (float*)d_out, V, NT, V, D, nullptr, nullptr, 1.0f, 0, 1);
}
